// CausalSelfAttention_8641474200362
// MI455X (gfx1250) — compile-verified, matrix operand order checked by reading, not run on hardware
//
#include <hip/hip_runtime.h>
#include <hip/hip_bf16.h>

// ---------------- problem constants ----------------
#define B_    2
#define T_    2048
#define C_    2048
#define NH_   16
#define NKV_  4
#define REP_  4
#define HD_   128
#define KVD_  512
#define QKVC_ 3072            // C + 2*KVD
#define EPS_  1.1920929e-07f

typedef _Float16 h16;
typedef __attribute__((ext_vector_type(8)))  _Float16 v8h;
typedef __attribute__((ext_vector_type(16))) _Float16 v16h;
typedef __attribute__((ext_vector_type(8)))  float    v8f;

__device__ __forceinline__ v8f z8() {
  v8f z = {0.f,0.f,0.f,0.f,0.f,0.f,0.f,0.f};
  return z;
}

// Build a 16x32 f16 A/B fragment for this lane.
// Per ISA layout: lane l supplies row (l&15); K halves at kb=(l>>4)*8 and kb+16.
// Caller passes p = row_base + kb (16B aligned).
__device__ __forceinline__ v16h frag_ld(const h16* p) {
  v8h lo = *(const v8h*)(p);
  v8h hi = *(const v8h*)(p + 16);
  v16h f;
#pragma unroll
  for (int i = 0; i < 8; ++i) { f[i] = lo[i]; f[i + 8] = hi[i]; }
  return f;
}

__device__ __forceinline__ v8f wmma16(v16h a, v16h b, v8f c) {
  return __builtin_amdgcn_wmma_f32_16x16x32_f16(
      /*neg_a=*/false, a, /*neg_b=*/false, b,
      /*c_mod=*/(short)0, c, /*reuse_a=*/false, /*reuse_b=*/false);
}

// CDNA5 async global->LDS copy (ASYNCcnt-tracked, GVS addressing).
// lds_addr: wave-relative LDS byte address (low 32 bits of generic pointer)
// gbase:    uniform 64-bit base (SGPR pair), goff: per-lane byte offset
__device__ __forceinline__ void async_b128(uint32_t lds_addr, uint64_t gbase, uint32_t goff) {
  asm volatile("global_load_async_to_lds_b128 %0, %1, %2"
               :: "v"(lds_addr), "v"(goff), "s"(gbase) : "memory");
}
#define WAIT_ASYNC(n) asm volatile("s_wait_asynccnt " #n ::: "memory")

// ---------------- K1: transpose + f32->f16 convert of weights ----------------
// in: [K][N] f32 row-major;  out: [N][K] f16 row-major
__global__ __launch_bounds__(256)
void transpose_w(const float* __restrict__ in, h16* __restrict__ out, int K, int N) {
  size_t idx = (size_t)blockIdx.x * 256 + threadIdx.x;
  size_t total = (size_t)K * N;
  if (idx >= total) return;
  int n = (int)(idx % N);
  int k = (int)(idx / N);
  out[(size_t)n * K + k] = (h16)in[idx];
}

// ---------------- K2: row RMSNorm of x, f32 -> f16 ----------------
__global__ __launch_bounds__(256)
void rmsnorm_rows(const float* __restrict__ x, h16* __restrict__ xn, int cols) {
  __shared__ float red[8];
  int row = blockIdx.x;
  const float* xr = x + (size_t)row * cols;
  float ss = 0.f;
  for (int c = threadIdx.x; c < cols; c += 256) { float v = xr[c]; ss += v * v; }
#pragma unroll
  for (int m = 16; m >= 1; m >>= 1) ss += __shfl_xor(ss, m, 32);
  if ((threadIdx.x & 31) == 0) red[threadIdx.x >> 5] = ss;
  __syncthreads();
  float tot = 0.f;
#pragma unroll
  for (int i = 0; i < 8; ++i) tot += red[i];
  float rs = rsqrtf(tot / (float)cols + EPS_);
  for (int c = threadIdx.x; c < cols; c += 256)
    xn[(size_t)row * cols + c] = (h16)(xr[c] * rs);
}

// ---------- K3/K6: 256x128 WMMA GEMM, async double-buffered LDS pipeline ----------
// C = A[MxK] * Bt[NxK]^T, f16 in, f32 accum.
// 8 waves as 4(m) x 2(n); each wave computes 64x64 via 4x4 accumulator fragments.
// K-tile = 64: per tile each wave does 16 frag loads and 32 WMMAs.
template<bool OUT_F32>
__global__ __launch_bounds__(256)
void gemm_async(const h16* __restrict__ A, const h16* __restrict__ Bt,
                void* __restrict__ C, int M, int N, int K) {
  (void)M;
  constexpr int LS = 72;                 // f16 row stride: 144B (16B-aligned frags)
  __shared__ h16 As[2][256 * LS];
  __shared__ h16 Bs[2][128 * LS];
  const int tid  = threadIdx.x;
  const int lane = tid & 31;
  const int wave = tid >> 5;
  const int m0 = blockIdx.y * 256;
  const int n0 = blockIdx.x * 128;
  const int wm = (wave & 3) * 64;
  const int wn = (wave >> 2) * 64;
  const int lrow = lane & 15;
  const int kb   = (lane >> 4) * 8;

  const uint64_t aBase = (uint64_t)(size_t)A;
  const uint64_t bBase = (uint64_t)(size_t)Bt;
  const uint32_t asLds0 = (uint32_t)(size_t)&As[0][0];
  const uint32_t asLds1 = (uint32_t)(size_t)&As[1][0];
  const uint32_t bsLds0 = (uint32_t)(size_t)&Bs[0][0];
  const uint32_t bsLds1 = (uint32_t)(size_t)&Bs[1][0];

  // Issue one K-tile worth of async copies: A 256x64 (8 chunks/thread),
  // B 128x64 (4 chunks/thread); 16B per chunk.
  auto issue = [&](int buf, int k0) {
    uint32_t aL = buf ? asLds1 : asLds0;
    uint32_t bL = buf ? bsLds1 : bsLds0;
#pragma unroll
    for (int it = 0; it < 8; ++it) {
      int c   = it * 256 + tid;                 // 0..2047
      int row = c >> 3;
      int cb  = (c & 7) * 16;                   // byte offset inside 128B K-slab
      async_b128(aL + (uint32_t)(row * (LS * 2) + cb),
                 aBase, (uint32_t)(((m0 + row) * K + k0) * 2 + cb));
    }
#pragma unroll
    for (int it = 0; it < 4; ++it) {
      int c   = it * 256 + tid;                 // 0..1023
      int row = c >> 3;
      int cb  = (c & 7) * 16;
      async_b128(bL + (uint32_t)(row * (LS * 2) + cb),
                 bBase, (uint32_t)(((n0 + row) * K + k0) * 2 + cb));
    }
  };

  v8f acc[4][4];
#pragma unroll
  for (int i = 0; i < 4; ++i)
#pragma unroll
    for (int j = 0; j < 4; ++j) acc[i][j] = z8();

  const int nt = K / 64;
  issue(0, 0);

  for (int t = 0; t < nt; ++t) {
    const int cur = t & 1;
    if (t + 1 < nt) {
      issue(cur ^ 1, (t + 1) * 64);
      WAIT_ASYNC(12);        // in-order completion: tile t has landed
    } else {
      WAIT_ASYNC(0);
    }
    __syncthreads();
    const h16* as = &As[cur][0];
    const h16* bs = &Bs[cur][0];
#pragma unroll
    for (int ks = 0; ks < 2; ++ks) {
      v16h af[4], bf[4];
#pragma unroll
      for (int i = 0; i < 4; ++i)
        af[i] = frag_ld(&as[(wm + i * 16 + lrow) * LS + ks * 32 + kb]);
#pragma unroll
      for (int j = 0; j < 4; ++j)
        bf[j] = frag_ld(&bs[(wn + j * 16 + lrow) * LS + ks * 32 + kb]);
#pragma unroll
      for (int i = 0; i < 4; ++i)
#pragma unroll
        for (int j = 0; j < 4; ++j)
          acc[i][j] = wmma16(af[i], bf[j], acc[i][j]);
    }
    __syncthreads();
  }

  // C/D fragment layout: VGPR r -> (m = r + 8*(lane>=16), n = lane&15)
  float* Cf = (float*)C;
  h16*   Ch = (h16*)C;
  const int mh = (lane >> 4) * 8;
#pragma unroll
  for (int i = 0; i < 4; ++i) {
#pragma unroll
    for (int j = 0; j < 4; ++j) {
      int n = n0 + wn + j * 16 + lrow;
#pragma unroll
      for (int r = 0; r < 8; ++r) {
        int m = m0 + wm + i * 16 + mh + r;
        if constexpr (OUT_F32) Cf[(size_t)m * N + n] = acc[i][j][r];
        else                   Ch[(size_t)m * N + n] = (h16)acc[i][j][r];
      }
    }
  }
}

// ---------------- K4: per-(token,head) RMSNorm + RoPE + gain, relayout ----------------
// grid: (B*T, 24) ; heads 0..15 = Q, 16..19 = K, 20..23 = V ; 128 threads = HD
__global__ __launch_bounds__(128)
void qkv_post(const h16* __restrict__ qkv, const float* __restrict__ q_gain,
              h16* __restrict__ q_h, h16* __restrict__ k_h, h16* __restrict__ v_t) {
  const int bt   = blockIdx.x;
  const int head = blockIdx.y;
  const int b = bt / T_, t = bt % T_;
  const int d = threadIdx.x;
  const h16* row = qkv + (size_t)bt * QKVC_;

  int col;
  if (head < 16)      col = head * HD_ + d;
  else if (head < 20) col = C_   + (head - 16) * HD_ + d;
  else                col = C_ + KVD_ + (head - 20) * HD_ + d;
  float v = (float)row[col];

  if (head >= 20) {       // V: transpose to [B][NKV][HD][T]
    int g = head - 20;
    v_t[(((size_t)b * NKV_ + g) * HD_ + d) * T_ + t] = (h16)v;
    return;
  }

  __shared__ float vals[HD_];
  __shared__ float red[4];
  vals[d] = v;
  float ss = v * v;
#pragma unroll
  for (int m = 16; m >= 1; m >>= 1) ss += __shfl_xor(ss, m, 32);
  if ((d & 31) == 0) red[d >> 5] = ss;
  __syncthreads();
  float tot = red[0] + red[1] + red[2] + red[3];
  float rs = rsqrtf(tot / (float)HD_ + EPS_);

  float xn    = v * rs;
  float other = vals[d < 64 ? d + 64 : d - 64] * rs;
  float rot   = (d < 64) ? -other : other;
  int   i     = d & 63;
  // inv_freq = 10000^(-i/64) = exp(-i * ln(10000)/64)
  float ang = (float)t * __expf(-(float)i * (9.210340371976184f / 64.f));
  float cs = __cosf(ang), sn = __sinf(ang);
  float rres = xn * cs + rot * sn;

  if (head < 16) {
    rres *= q_gain[head];
    int g = head >> 2, rr = head & 3;
    q_h[((((size_t)b * NKV_ + g) * REP_ + rr) * T_ + t) * HD_ + d] = (h16)rres;
  } else {
    int g = head - 16;
    k_h[(((size_t)b * NKV_ + g) * T_ + t) * HD_ + d] = (h16)rres;
  }
}

// ---------------- K5: causal flash attention, WMMA scores + PV ----------------
// grid: (T/64, NKV*REP, B); 128 threads = 4 waves, each wave owns 16 query rows.
__global__ __launch_bounds__(128)
void flash_attn(const h16* __restrict__ q_h, const h16* __restrict__ k_h,
                const h16* __restrict__ v_t, h16* __restrict__ o) {
  const int qt = blockIdx.x;
  const int g  = blockIdx.y / REP_;
  const int r  = blockIdx.y % REP_;
  const int b  = blockIdx.z;
  const int wave = threadIdx.x >> 5;
  const int lane = threadIdx.x & 31;
  const int lrow = lane & 15;
  const int kb   = (lane >> 4) * 8;
  const int mh   = (lane >> 4) * 8;
  const int q0   = qt * 64 + wave * 16;

  const h16* qbase = q_h + ((((size_t)b * NKV_ + g) * REP_ + r) * T_) * HD_;
  const h16* kbase = k_h + (((size_t)b * NKV_ + g) * T_) * HD_;
  const h16* vbase = v_t + (((size_t)b * NKV_ + g) * HD_) * T_;

  // Persistent Q fragments: 16 rows x 128 K, 4 K-steps of 32
  v16h qf[4];
#pragma unroll
  for (int ks = 0; ks < 4; ++ks)
    qf[ks] = frag_ld(qbase + (size_t)(q0 + lrow) * HD_ + ks * 32 + kb);

  v8f oacc[8];
#pragma unroll
  for (int nf = 0; nf < 8; ++nf) oacc[nf] = z8();
  float rm[8], rl[8];
#pragma unroll
  for (int j = 0; j < 8; ++j) { rm[j] = -1e30f; rl[j] = 0.f; }

  __shared__ h16 plds[4][16 * 40];     // per-wave P staging (C-frag -> A-frag)
  h16* pw = plds[wave];

  const float scale = 0.08838834764831845f;   // 1/sqrt(128)
  const int ntiles = (q0 + 15) / 32 + 1;      // causal bound for this wave

  for (int kt = 0; kt < ntiles; ++kt) {
    const int key0 = kt * 32;
    v8f s0 = z8(), s1 = z8();
#pragma unroll
    for (int ks = 0; ks < 4; ++ks) {
      v16h kf0 = frag_ld(kbase + (size_t)(key0 +      lrow) * HD_ + ks * 32 + kb);
      v16h kf1 = frag_ld(kbase + (size_t)(key0 + 16 + lrow) * HD_ + ks * 32 + kb);
      s0 = wmma16(qf[ks], kf0, s0);
      s1 = wmma16(qf[ks], kf1, s1);
    }
    // causal mask + online softmax (row reductions stay inside 16-lane groups)
#pragma unroll
    for (int j = 0; j < 8; ++j) {
      int qrow  = q0 + mh + j;
      int key_a = key0 + lrow;
      int key_b = key0 + 16 + lrow;
      float a  = s0[j] * scale;
      float bb = s1[j] * scale;
      if (key_a > qrow) a  = -1e30f;
      if (key_b > qrow) bb = -1e30f;
      float mx = fmaxf(a, bb);
#pragma unroll
      for (int m = 8; m >= 1; m >>= 1) mx = fmaxf(mx, __shfl_xor(mx, m, 32));
      float nm = fmaxf(rm[j], mx);
      float alpha = __expf(rm[j] - nm);
      float pa = __expf(a  - nm);
      float pb = __expf(bb - nm);
      float sum = pa + pb;
#pragma unroll
      for (int m = 8; m >= 1; m >>= 1) sum += __shfl_xor(sum, m, 32);
      rl[j] = rl[j] * alpha + sum;
      rm[j] = nm;
#pragma unroll
      for (int nf = 0; nf < 8; ++nf) oacc[nf][j] *= alpha;
      pw[(mh + j) * 40 + lrow]      = (h16)pa;
      pw[(mh + j) * 40 + 16 + lrow] = (h16)pb;
    }
    // Restage P as an A fragment (same-wave LDS dependency; DScnt-ordered)
    v16h pf = frag_ld(&pw[lrow * 40 + kb]);
#pragma unroll
    for (int nf = 0; nf < 8; ++nf) {
      v16h vf = frag_ld(vbase + (size_t)(nf * 16 + lrow) * T_ + key0 + kb);
      oacc[nf] = wmma16(pf, vf, oacc[nf]);
    }
  }

  // epilogue: normalize and write [B][T][C] head-block
  const size_t obase = (size_t)b * T_ * C_;
  const int c0 = (g * REP_ + r) * HD_;
#pragma unroll
  for (int j = 0; j < 8; ++j) {
    float inv = 1.f / rl[j];
    int t = q0 + mh + j;
#pragma unroll
    for (int nf = 0; nf < 8; ++nf)
      o[obase + (size_t)t * C_ + c0 + nf * 16 + lrow] = (h16)(oacc[nf][j] * inv);
  }
}

// ---------------- host launcher ----------------
extern "C" void kernel_launch(void* const* d_in, const int* in_sizes, int n_in,
                              void* d_out, int out_size, void* d_ws, size_t ws_size,
                              hipStream_t stream) {
  (void)in_sizes; (void)n_in; (void)out_size; (void)ws_size;
  const float* x      = (const float*)d_in[0];
  const float* w_qkv  = (const float*)d_in[1];
  const float* w_proj = (const float*)d_in[2];
  const float* q_gain = (const float*)d_in[3];
  float* out = (float*)d_out;

  // workspace carve-up (all f16)
  char* ws = (char*)d_ws;
  h16* w_qkv_t  = (h16*)ws;  ws += (size_t)QKVC_ * C_ * sizeof(h16);          // [3072][2048]
  h16* w_proj_t = (h16*)ws;  ws += (size_t)C_ * C_ * sizeof(h16);             // [2048][2048]
  h16* xn       = (h16*)ws;  ws += (size_t)B_ * T_ * C_ * sizeof(h16);        // [4096][2048]
  h16* qkv      = (h16*)ws;  ws += (size_t)B_ * T_ * QKVC_ * sizeof(h16);     // [4096][3072]
  h16* q_h      = (h16*)ws;  ws += (size_t)B_ * NKV_ * REP_ * T_ * HD_ * sizeof(h16);
  h16* k_h      = (h16*)ws;  ws += (size_t)B_ * NKV_ * T_ * HD_ * sizeof(h16);
  h16* v_t      = (h16*)ws;  ws += (size_t)B_ * NKV_ * HD_ * T_ * sizeof(h16);
  h16* attn_o   = (h16*)ws;  ws += (size_t)B_ * T_ * C_ * sizeof(h16);

  const int BT = B_ * T_;

  // 1) weight transpose + convert
  {
    size_t tot = (size_t)C_ * QKVC_;
    transpose_w<<<dim3((tot + 255) / 256), dim3(256), 0, stream>>>(w_qkv, w_qkv_t, C_, QKVC_);
  }
  {
    size_t tot = (size_t)C_ * C_;
    transpose_w<<<dim3((tot + 255) / 256), dim3(256), 0, stream>>>(w_proj, w_proj_t, C_, C_);
  }

  // 2) input RMSNorm
  rmsnorm_rows<<<dim3(BT), dim3(256), 0, stream>>>(x, xn, C_);

  // 3) QKV GEMM: [4096 x 2048] * [2048 x 3072] -> f16
  gemm_async<false><<<dim3(QKVC_ / 128, BT / 256), dim3(256), 0, stream>>>(
      xn, w_qkv_t, (void*)qkv, BT, QKVC_, C_);

  // 4) head RMSNorm + RoPE + gain + relayout
  qkv_post<<<dim3(BT, NH_ + 2 * NKV_), dim3(128), 0, stream>>>(
      qkv, q_gain, q_h, k_h, v_t);

  // 5) causal flash attention
  flash_attn<<<dim3(T_ / 64, NKV_ * REP_, B_), dim3(128), 0, stream>>>(
      q_h, k_h, v_t, attn_o);

  // 6) output projection: [4096 x 2048] * [2048 x 2048] -> f32 d_out
  gemm_async<true><<<dim3(C_ / 128, BT / 256), dim3(256), 0, stream>>>(
      attn_o, w_proj_t, (void*)out, BT, C_, C_);
}
